// BaseDecoder_82025285419169
// MI455X (gfx1250) — compile-verified, matrix operand order checked by reading, not run on hardware
//
#include <hip/hip_runtime.h>
#include <math.h>

typedef __attribute__((ext_vector_type(16))) _Float16 v16h;
typedef __attribute__((ext_vector_type(8)))  _Float16 v8h;
typedef __attribute__((ext_vector_type(4)))  _Float16 v4h;
typedef __attribute__((ext_vector_type(8)))  float    v8f;
typedef __attribute__((ext_vector_type(4)))  float    v4f;

#define MT   128           // pixels per block tile
#define NT   64            // output channels per block tile
#define KT   64            // K elements staged per step (2 WMMA K-slices)
#define LDSK 72            // padded K stride (halves): 144B rows -> conflict-free frags

#define CAT16(lo, hi) __builtin_shufflevector((lo), (hi), 0, 1, 2, 3, 4, 5, 6, 7, \
                                              8, 9, 10, 11, 12, 13, 14, 15)

// ---------------------------------------------------------------------------
// Per-layer weight convert: f32 [Cout][Ktot] -> f16 [Cpad][KTpad], zero padded.
// Lets the conv kernel load B fragments straight from global (cached in WGP$).
// ---------------------------------------------------------------------------
__global__ void wcvt_kernel(const float* __restrict__ w, _Float16* __restrict__ o,
                            int Cout, int Ktot, int Cpad, int KTpad)
{
    const int t = blockIdx.x * blockDim.x + threadIdx.x;
    if (t >= Cpad * KTpad) return;
    const int k = t % KTpad;
    const int c = t / KTpad;
    const float v = (c < Cout && k < Ktot) ? w[(long)c * Ktot + k] : 0.f;
    o[t] = (_Float16)v;
}

// ---------------------------------------------------------------------------
// Implicit-GEMM conv (KS=3 'SAME' zero-pad, or KS=1) with optional 2-tensor
// channel-concat input, fused bias + activation (0=none,1=relu,2=prelu).
// GEMM: M = B*H*W pixels, N = Cout, K = Cin*KS*KS.  f16 WMMA, f32 accum.
// A tile staged via LDS (im2col + f32->f16); B fragments loaded directly from
// the pre-converted f16 weight buffer.  Block: 8 waves as 4(M) x 2(N); each
// wave owns a 32x32 tile -> 8 v_wmma per 64-K step.
// ---------------------------------------------------------------------------
template <int KS>
__launch_bounds__(256)
__global__ void conv_wmma_kernel(const float* __restrict__ in1,
                                 const float* __restrict__ in2,
                                 int Cin1, int Cin, int Bn, int H, int W,
                                 const _Float16* __restrict__ wh, // [Cpad][KTpad] f16
                                 int KTpad,
                                 const float* __restrict__ bias,  // [Cout] or null
                                 const float* __restrict__ alpha, // prelu [Cout] or null
                                 float* __restrict__ out,         // [Bn][Cout][H][W]
                                 int Cout, int act)
{
    constexpr int KS2 = KS * KS;
    constexpr int PAD = KS / 2;

    __shared__ __align__(16) _Float16 Ah[MT][LDSK];

    const int tid   = threadIdx.x;
    const int lane  = tid & 31;
    const int wave  = tid >> 5;
    const int waveM = wave >> 1;      // 0..3 -> 32-pixel sub-block
    const int waveN = wave & 1;       // 0..1 -> 32-channel sub-block
    const int HW    = H * W;
    const int P     = Bn * HW;
    const int Ktot  = Cin * KS2;
    const int Cin2  = Cin - Cin1;

    const int pBase = blockIdx.x * MT;
    const int nBase = blockIdx.y * NT;

    // ---- A staging map: row m = tid&127, K half-range = (tid>>7)*32 ----
    const int  am    = tid & 127;
    const int  akblk = (tid >> 7) * 32;   // 0 or 32
    const int  ap    = pBase + am;
    const bool apv   = ap < P;
    int apn = 0, ay = 0, ax = 0;
    if (apv) {
        apn = ap / HW;
        const int s = ap - apn * HW;
        ay = s / W;
        ax = s - ay * W;
    }

    // ---- per-lane fragment geometry ----
    const int hl = lane & 15;
    const int kq = (lane >> 4) * 8;       // lanes 16-31: K 8..15 / 24..31

    // B fragment row pointers (padded f16 weights, hoisted)
    const _Float16* wrow0 = wh + (long)(nBase + waveN * 32 + hl) * KTpad;
    const _Float16* wrow1 = wh + (long)(nBase + waveN * 32 + 16 + hl) * KTpad;

    v8f acc[2][2];
    #pragma unroll
    for (int mi = 0; mi < 2; ++mi)
        #pragma unroll
        for (int ni = 0; ni < 2; ++ni)
            acc[mi][ni] = v8f{0.f, 0.f, 0.f, 0.f, 0.f, 0.f, 0.f, 0.f};

    for (int kk = 0; kk < Ktot; kk += KT) {
        __syncthreads();
        // ------- stage A tile: 128 pixels x 64 K, packed b64 stores -------
        #pragma unroll
        for (int g = 0; g < 8; ++g) {
            const int k0 = akblk + g * 4;
            v4h pk;
            #pragma unroll
            for (int e = 0; e < 4; ++e) {
                const int kg = kk + k0 + e;
                float v = 0.f;
                if (apv && kg < Ktot) {
                    int ci, yy, xx;
                    if (KS == 1) {
                        ci = kg; yy = ay; xx = ax;
                    } else {
                        ci = kg / KS2;                   // compile-time divisor
                        const int r  = kg - ci * KS2;
                        const int ry = r / KS;
                        yy = ay + ry - PAD;
                        xx = ax + (r - ry * KS) - PAD;
                    }
                    if (yy >= 0 && yy < H && xx >= 0 && xx < W) {
                        const float* src = (ci < Cin1)
                            ? in1 + ((long)apn * Cin1 + ci) * HW
                            : in2 + ((long)apn * Cin2 + (ci - Cin1)) * HW;
                        v = src[yy * W + xx];
                    }
                }
                pk[e] = (_Float16)v;
            }
            *(v4h*)&Ah[am][k0] = pk;
        }
        __syncthreads();

        // ------- fragments; 8 WMMAs per step -------
        #pragma unroll
        for (int s = 0; s < 2; ++s) {         // two K=32 slices
            const int kb = s * 32 + kq;
            const int ko = kk + kb;

            const _Float16* ar0 = &Ah[waveM * 32 + hl][kb];
            const _Float16* ar1 = &Ah[waveM * 32 + 16 + hl][kb];
            const v16h af0 = CAT16(*(const v8h*)ar0, *(const v8h*)(ar0 + 16));
            const v16h af1 = CAT16(*(const v8h*)ar1, *(const v8h*)(ar1 + 16));
            const v16h bf0 = CAT16(*(const v8h*)(wrow0 + ko),
                                   *(const v8h*)(wrow0 + ko + 16));
            const v16h bf1 = CAT16(*(const v8h*)(wrow1 + ko),
                                   *(const v8h*)(wrow1 + ko + 16));

            acc[0][0] = __builtin_amdgcn_wmma_f32_16x16x32_f16(
                false, af0, false, bf0, (short)0, acc[0][0], false, false);
            acc[0][1] = __builtin_amdgcn_wmma_f32_16x16x32_f16(
                false, af0, false, bf1, (short)0, acc[0][1], false, false);
            acc[1][0] = __builtin_amdgcn_wmma_f32_16x16x32_f16(
                false, af1, false, bf0, (short)0, acc[1][0], false, false);
            acc[1][1] = __builtin_amdgcn_wmma_f32_16x16x32_f16(
                false, af1, false, bf1, (short)0, acc[1][1], false, false);
        }
    }

    // ------- epilogue: lane = one channel, 8 consecutive pixels, x4 tiles -------
    #pragma unroll
    for (int ni = 0; ni < 2; ++ni) {
        const int co = nBase + waveN * 32 + ni * 16 + hl;
        if (co >= Cout) continue;
        const float bv = bias ? bias[co] : 0.f;
        const float al = (act == 2 && alpha) ? alpha[co] : 0.f;
        #pragma unroll
        for (int mi = 0; mi < 2; ++mi) {
            const int m0 = waveM * 32 + mi * 16 + (lane >> 4) * 8;
            const int p0 = pBase + m0;
            if (p0 >= P) continue;
            const int pn = p0 / HW;
            const int s0 = p0 - pn * HW;
            float* dst = out + ((long)pn * Cout + co) * HW + s0;

            float r[8];
            #pragma unroll
            for (int j = 0; j < 8; ++j) {
                float v = acc[mi][ni][j] + bv;
                if (act == 1)      v = fmaxf(v, 0.f);
                else if (act == 2) v = (v >= 0.f) ? v : al * v;
                r[j] = v;
            }
            if (p0 + 7 < P) {
                // 16B-aligned (pBase % 128 == 0, HW % 8 == 0) -> two b128 stores
                *(v4f*)dst       = v4f{r[0], r[1], r[2], r[3]};
                *(v4f*)(dst + 4) = v4f{r[4], r[5], r[6], r[7]};
            } else {
                for (int j = 0; j < 8 && p0 + j < P; ++j) dst[j] = r[j];
            }
        }
    }
}

// ---------------------------------------------------------------------------
// 2x2 mean pool (adaptive avg pool to half resolution)
// ---------------------------------------------------------------------------
__global__ void pool2_kernel(const float* __restrict__ in, float* __restrict__ out,
                             int total, int Ho, int Wo)
{
    const int t = blockIdx.x * blockDim.x + threadIdx.x;
    if (t >= total) return;
    const int wo = t % Wo;
    const int ho = (t / Wo) % Ho;
    const int bc = t / (Wo * Ho);
    const int Wi = Wo * 2;
    const float* p = in + ((long)bc * (Ho * 2) + ho * 2) * Wi + wo * 2;
    out[t] = 0.25f * (p[0] + p[1] + p[Wi] + p[Wi + 1]);
}

// ---------------------------------------------------------------------------
// CAUN assembly for a channel chunk [c0, c0+Cg):
// softmax over 9 mask logits, edge-clamped 3x3 gather, faithful scrambled
// scatter: out[b,c, w*2+v, h*2+u] (transpose(0,1,5,3,4,2) of [B,C,u,v,h,w]).
// mask chunk layout: [B][Cg*36][H][W], cc = ((cl*9+n)*2+u)*2+v.
// ---------------------------------------------------------------------------
__global__ void caun_assemble_kernel(const float* __restrict__ xlow,  // [B][C][H][W]
                                     const float* __restrict__ mask,  // chunk
                                     float* __restrict__ out,         // [B][C][2H][2W]
                                     int C, int c0, int Cg, int H, int W, int total)
{
    const int t = blockIdx.x * blockDim.x + threadIdx.x;
    if (t >= total) return;
    const int HW = H * W;
    const int w  = t % W;
    const int h  = (t / W) % H;
    const int cl = (t / HW) % Cg;
    const int b  = t / (HW * Cg);
    const int c  = c0 + cl;

    const float* xp = xlow + ((long)b * C + c) * HW;
    float nb[9];
    #pragma unroll
    for (int i = 0; i < 3; ++i)
        #pragma unroll
        for (int j = 0; j < 3; ++j) {
            const int yy = min(max(h + i - 1, 0), H - 1);
            const int xx = min(max(w + j - 1, 0), W - 1);
            nb[i * 3 + j] = xp[yy * W + xx];
        }

    const float* mb = mask + ((long)b * (Cg * 36)) * HW + (long)h * W + w;
    float* ob = out + ((long)b * C + c) * (4L * HW);
    #pragma unroll
    for (int u = 0; u < 2; ++u)
        #pragma unroll
        for (int v = 0; v < 2; ++v) {
            float mv[9];
            float mx = -1e30f;
            #pragma unroll
            for (int n = 0; n < 9; ++n) {
                mv[n] = mb[((long)(((cl * 9 + n) * 2 + u) * 2 + v)) * HW];
                mx = fmaxf(mx, mv[n]);
            }
            float se = 0.f, av = 0.f;
            #pragma unroll
            for (int n = 0; n < 9; ++n) {
                const float e = __expf(mv[n] - mx);
                se += e;
                av += e * nb[n];
            }
            ob[(long)(w * 2 + v) * (2 * W) + (h * 2 + u)] = av / se;
        }
}

// ---------------------------------------------------------------------------
// Memory-bank cosine attention residual fused with final 1x1 conv (32->2).
// Two-pass softmax over the 64 slots (recompute dots) -> no lg[64] array,
// no register spills.  One thread per pixel.
// ---------------------------------------------------------------------------
__launch_bounds__(256)
__global__ void memfuse_kernel(const float* __restrict__ x03,  // [2][32][HW]
                               const float* __restrict__ mem,  // [64][32]
                               const float* __restrict__ finw, // [2][32]
                               const float* __restrict__ finb, // [2]
                               float* __restrict__ out,        // [2][2][HW]
                               int HW)
{
    __shared__ float mn[64][33];
    __shared__ float fw[64];
    const int t = threadIdx.x;
    if (t < 64) {
        float s = 0.f;
        #pragma unroll
        for (int c = 0; c < 32; ++c) { const float v = mem[t * 32 + c]; s += v * v; }
        const float inv = 1.f / fmaxf(sqrtf(s), 1e-12f);
        #pragma unroll
        for (int c = 0; c < 32; ++c) mn[t][c] = mem[t * 32 + c] * inv;
        fw[t] = finw[t];
    }
    __syncthreads();

    const int p = blockIdx.x * 256 + t;          // grid covers 2*HW exactly
    const int b = p / HW;
    const int s = p - b * HW;
    const float* xin = x03 + ((long)b * 32) * HW + s;

    float f[32];
    float nrm = 0.f;
    #pragma unroll
    for (int c = 0; c < 32; ++c) { f[c] = xin[(long)c * HW]; nrm += f[c] * f[c]; }
    const float inv = 1.f / fmaxf(sqrtf(nrm), 1e-12f);

    // pass 1: max logit
    float mx = -1e30f;
    for (int k = 0; k < 64; ++k) {
        float d = 0.f;
        #pragma unroll
        for (int c = 0; c < 32; ++c) d += f[c] * mn[k][c];
        mx = fmaxf(mx, d * inv);
    }
    // pass 2: exp-sum + weighted reconstruction
    float se = 0.f;
    float rec[32];
    #pragma unroll
    for (int c = 0; c < 32; ++c) rec[c] = 0.f;
    for (int k = 0; k < 64; ++k) {
        float d = 0.f;
        #pragma unroll
        for (int c = 0; c < 32; ++c) d += f[c] * mn[k][c];
        const float e = __expf(d * inv - mx);
        se += e;
        #pragma unroll
        for (int c = 0; c < 32; ++c) rec[c] += e * mn[k][c];
    }
    const float rinv = 1.f / se;

    #pragma unroll
    for (int k = 0; k < 2; ++k) {
        float o = finb[k];
        #pragma unroll
        for (int c = 0; c < 32; ++c) o += fw[k * 32 + c] * (f[c] + rec[c] * rinv);
        out[((long)b * 2 + k) * HW + s] = o;
    }
}

// ---------------------------------------------------------------------------
extern "C" void kernel_launch(void* const* d_in, const int* in_sizes, int n_in,
                              void* d_out, int out_size, void* d_ws, size_t ws_size,
                              hipStream_t stream)
{
    (void)in_sizes; (void)n_in; (void)out_size; (void)ws_size;
    const float* x0    = (const float*)d_in[0];
    const float* x1    = (const float*)d_in[1];
    const float* x2    = (const float*)d_in[2];
    const float* x3    = (const float*)d_in[3];
    const float* c3_w1 = (const float*)d_in[4],  *c3_b1 = (const float*)d_in[5],  *c3_a1 = (const float*)d_in[6];
    const float* c3_w2 = (const float*)d_in[7],  *c3_b2 = (const float*)d_in[8],  *c3_a2 = (const float*)d_in[9];
    const float* c3_kw = (const float*)d_in[10], *c3_kb = (const float*)d_in[11];
    const float* c2_w1 = (const float*)d_in[12], *c2_b1 = (const float*)d_in[13], *c2_a1 = (const float*)d_in[14];
    const float* c2_w2 = (const float*)d_in[15], *c2_b2 = (const float*)d_in[16], *c2_a2 = (const float*)d_in[17];
    const float* c2_kw = (const float*)d_in[18], *c2_kb = (const float*)d_in[19];
    const float* c1_w1 = (const float*)d_in[20], *c1_b1 = (const float*)d_in[21], *c1_a1 = (const float*)d_in[22];
    const float* c1_w2 = (const float*)d_in[23], *c1_b2 = (const float*)d_in[24], *c1_a2 = (const float*)d_in[25];
    const float* c1_kw = (const float*)d_in[26], *c1_kb = (const float*)d_in[27];
    const float* b21_w1 = (const float*)d_in[28], *b21_b1 = (const float*)d_in[29];
    const float* b21_w2 = (const float*)d_in[30], *b21_b2 = (const float*)d_in[31];
    const float* b12_w1 = (const float*)d_in[32], *b12_b1 = (const float*)d_in[33];
    const float* b12_w2 = (const float*)d_in[34], *b12_b2 = (const float*)d_in[35];
    const float* b03_w1 = (const float*)d_in[36], *b03_b1 = (const float*)d_in[37];
    const float* b03_w2 = (const float*)d_in[38], *b03_b2 = (const float*)d_in[39];
    const float* fin_w  = (const float*)d_in[40], *fin_b  = (const float*)d_in[41];
    const float* memb   = (const float*)d_in[42];

    // ---- workspace bump allocation (floats) ----
    float* ws = (float*)d_ws;
    size_t off = 0;
    auto A = [&](size_t n) { float* p = ws + off; off += n; return p; };
    float* x3up    = A(2L * 256 * 64 * 64);
    float* x21     = A(2L * 128 * 64 * 64);
    float* x2up    = A(2L * 128 * 128 * 128);
    float* x12     = A(2L * 64 * 128 * 128);
    float* x1up    = A(2L * 64 * 256 * 256);
    float* x03     = A(2L * 32 * 256 * 256);
    float* tmp     = A(2L * 32 * 256 * 256);       // block-conv intermediates (max)
    float* ctxb    = A(2L * 32 * 128 * 128);       // pooled ctx (max level)
    float* fa      = A(2L * 32 * 128 * 128);       // caun feature a (max)
    float* fb      = A(2L * 16 * 128 * 128);       // caun feature b (max)
    float* maskbuf = A(2L * (8 * 36) * 128 * 128); // largest mask chunk
    _Float16* wh   = (_Float16*)A(262144);         // 524288 halves (max 442368 used)

    auto conv = [&](const float* i1, const float* i2, int Cin1, int Cin,
                    int H, int W, const float* w, const float* b, const float* al,
                    float* o, int Cout, int ks, int act) {
        const int Ktot  = Cin * ks * ks;
        const int KTpad = (Ktot + 63) & ~63;
        const int gridN = (Cout + NT - 1) / NT;
        const int Cpad  = gridN * NT;
        const int wtot  = Cpad * KTpad;
        wcvt_kernel<<<dim3((wtot + 255) / 256), 256, 0, stream>>>(w, wh, Cout, Ktot,
                                                                  Cpad, KTpad);
        const int P = 2 * H * W;
        dim3 grid((P + MT - 1) / MT, gridN);
        if (ks == 3)
            conv_wmma_kernel<3><<<grid, 256, 0, stream>>>(i1, i2, Cin1, Cin, 2, H, W,
                                                          wh, KTpad, b, al, o, Cout, act);
        else
            conv_wmma_kernel<1><<<grid, 256, 0, stream>>>(i1, i2, Cin1, Cin, 2, H, W,
                                                          wh, KTpad, b, al, o, Cout, act);
    };
    auto pool = [&](const float* in, float* o, int C, int Ho, int Wo) {
        const int total = 2 * C * Ho * Wo;
        pool2_kernel<<<dim3((total + 255) / 256), 256, 0, stream>>>(in, o, total, Ho, Wo);
    };
    auto caun = [&](const float* xlow, const float* f, int Cf,
                    const float* kw, const float* kb,
                    float* o, int C, int Cg, int H, int W) {
        for (int c0 = 0; c0 < C; c0 += Cg) {
            conv(f, nullptr, Cf, Cf, H, W, kw + (long)c0 * 36 * Cf, kb + c0 * 36,
                 nullptr, maskbuf, Cg * 36, 1, 0);
            const int total = 2 * Cg * H * W;
            caun_assemble_kernel<<<dim3((total + 255) / 256), 256, 0, stream>>>(
                xlow, maskbuf, o, C, c0, Cg, H, W, total);
        }
    };

    // ---------- caun3: x3_0 [2,256,32,32], ctx = pool(x2_0) ----------
    pool(x2, ctxb, 128, 32, 32);
    conv(ctxb, nullptr, 128, 128, 32, 32, c3_w1, c3_b1, c3_a1, fa, 128, 3, 2);
    conv(fa,   nullptr, 128, 128, 32, 32, c3_w2, c3_b2, c3_a2, fb, 64,  3, 2);
    caun(x3, fb, 64, c3_kw, c3_kb, x3up, 256, 32, 32, 32);

    // ---------- x2_1 = block(concat[x2_0, x3up]) ----------
    conv(x2,  x3up,    128, 384, 64, 64, b21_w1, b21_b1, nullptr, tmp, 128, 3, 1);
    conv(tmp, nullptr, 128, 128, 64, 64, b21_w2, b21_b2, nullptr, x21, 128, 3, 1);

    // ---------- caun2: x2_1, ctx = pool(x1_0) ----------
    pool(x1, ctxb, 64, 64, 64);
    conv(ctxb, nullptr, 64, 64, 64, 64, c2_w1, c2_b1, c2_a1, fa, 64, 3, 2);
    conv(fa,   nullptr, 64, 64, 64, 64, c2_w2, c2_b2, c2_a2, fb, 32, 3, 2);
    caun(x21, fb, 32, c2_kw, c2_kb, x2up, 128, 16, 64, 64);

    // ---------- x1_2 = block(concat[x1_0, x2up]) ----------
    conv(x1,  x2up,    64, 192, 128, 128, b12_w1, b12_b1, nullptr, tmp, 64, 3, 1);
    conv(tmp, nullptr, 64, 64,  128, 128, b12_w2, b12_b2, nullptr, x12, 64, 3, 1);

    // ---------- caun1: x1_2, ctx = pool(x0_0) ----------
    pool(x0, ctxb, 32, 128, 128);
    conv(ctxb, nullptr, 32, 32, 128, 128, c1_w1, c1_b1, c1_a1, fa, 32, 3, 2);
    conv(fa,   nullptr, 32, 32, 128, 128, c1_w2, c1_b2, c1_a2, fb, 16, 3, 2);
    caun(x12, fb, 16, c1_kw, c1_kb, x1up, 64, 8, 128, 128);

    // ---------- x0_3 = block(concat[x0_0, x1up]) ----------
    conv(x0,  x1up,    32, 96, 256, 256, b03_w1, b03_b1, nullptr, tmp, 32, 3, 1);
    conv(tmp, nullptr, 32, 32, 256, 256, b03_w2, b03_b2, nullptr, x03, 32, 3, 1);

    // ---------- memory-bank residual + final 1x1 conv ----------
    memfuse_kernel<<<dim3((2 * 65536) / 256), 256, 0, stream>>>(
        x03, memb, fin_w, fin_b, (float*)d_out, 65536);
}
